// HyperbolicGATConv_50792283242938
// MI455X (gfx1250) — matrix operand order repaired and checked by reading, not compiled, not run on hardware
//
#include <hip/hip_runtime.h>
#include <hip/hip_fp16.h>

typedef __attribute__((ext_vector_type(16))) _Float16 v16h;
typedef __attribute__((ext_vector_type(8)))  _Float16 v8h;
typedef __attribute__((ext_vector_type(8)))  float    v8f;

#define IN_DIM   128
#define OUT_DIM  64
#define NEG_SLOPE 0.2f
#define EPS 1e-7f

// ---- float <-> order-preserving int encoding (for exact float atomicMax) ----
__device__ __forceinline__ int f2ord(float f) {
  int i = __float_as_int(f);
  return i >= 0 ? i : (i ^ 0x7fffffff);
}
__device__ __forceinline__ float ord2f(int i) {
  return __int_as_float(i >= 0 ? i : (i ^ 0x7fffffff));
}

__device__ __forceinline__ float wave_sum32(float v) {
#pragma unroll
  for (int off = 16; off; off >>= 1) v += __shfl_xor(v, off, 32);
  return v;
}

// ============================================================================
// K1: u = logmap0(x) -> f16 [N,128]; also transpose W (f32 [128,64]) -> Wt f16 [64,128]
// wave-per-row: 32 lanes x float4 = 128 elements
// ============================================================================
__global__ void k_prep(const float* __restrict__ x, const float* __restrict__ W,
                       _Float16* __restrict__ A16, _Float16* __restrict__ Wt,
                       int nRows) {
  const int lane   = threadIdx.x & 31;
  const int wid    = blockIdx.x * (blockDim.x >> 5) + (threadIdx.x >> 5);
  const int nWaves = gridDim.x * (blockDim.x >> 5);

  // one-shot W transpose/convert by the first 8192 global threads
  int gtid = blockIdx.x * blockDim.x + threadIdx.x;
  if (gtid < IN_DIM * OUT_DIM) {
    int k = gtid >> 6, n = gtid & 63;
    Wt[n * IN_DIM + k] = (_Float16)W[k * OUT_DIM + n];
  }

  for (int row = wid; row < nRows; row += nWaves) {
    const float4 v = ((const float4*)(x + (size_t)row * IN_DIM))[lane];
    float ss = v.x * v.x + v.y * v.y + v.z * v.z + v.w * v.w;
    ss = wave_sum32(ss);
    float n = sqrtf(ss);
    n = fminf(fmaxf(n, EPS), 1.0f - 1e-5f);
    float scale = atanhf(n) / n;
    _Float16* ar = A16 + (size_t)row * IN_DIM + lane * 4;
    ar[0] = (_Float16)(v.x * scale);
    ar[1] = (_Float16)(v.y * scale);
    ar[2] = (_Float16)(v.z * scale);
    ar[3] = (_Float16)(v.w * scale);
  }
}

// ============================================================================
// K2: H = A16 @ W + b  via v_wmma_f32_16x16x32_f16. Wave computes 16x64 strip.
// ============================================================================
__global__ void k_gemm_wmma(const _Float16* __restrict__ A16,
                            const _Float16* __restrict__ Wt,   // [64][128]
                            const float* __restrict__ bias,
                            float* __restrict__ H, int nRows) {
  const int lane    = threadIdx.x & 31;
  const int half_id = lane >> 4;   // 0: lanes 0-15, 1: lanes 16-31
  const int r16     = lane & 15;
  const int wid     = blockIdx.x * (blockDim.x >> 5) + (threadIdx.x >> 5);
  const int m0      = wid * 16;
  if (m0 >= nRows) return;         // wave-uniform: EXEC all-ones inside

  v8f acc[4] = {v8f{}, v8f{}, v8f{}, v8f{}};

#pragma unroll
  for (int kb = 0; kb < IN_DIM; kb += 32) {
    // A fragment: row M = r16; K = kb + 8*half_id + [0..7]  and  kb + 16 + 8*half_id + [0..7]
    const _Float16* arow = A16 + (size_t)(m0 + r16) * IN_DIM + kb;
    v8h alo = *(const v8h*)(arow + 8 * half_id);
    v8h ahi = *(const v8h*)(arow + 16 + 8 * half_id);
    v16h afrag = __builtin_shufflevector(alo, ahi, 0,1,2,3,4,5,6,7,8,9,10,11,12,13,14,15);

#pragma unroll
    for (int nt = 0; nt < 4; ++nt) {
      // B fragment: column N = nt*16 + r16; K = kb + 8*half_id + [0..7]  and  kb + 16 + 8*half_id + [0..7]
      const _Float16* bcol = Wt + (size_t)(nt * 16 + r16) * IN_DIM + kb + 8 * half_id;
      v8h blo = *(const v8h*)(bcol);
      v8h bhi = *(const v8h*)(bcol + 16);
      v16h bfrag = __builtin_shufflevector(blo, bhi, 0,1,2,3,4,5,6,7,8,9,10,11,12,13,14,15);
      acc[nt] = __builtin_amdgcn_wmma_f32_16x16x32_f16(
          false, afrag, false, bfrag, (short)0, acc[nt], false, false);
    }
  }

  // D layout: element (M,N): lane = N + 16*(M/8) within tile, VGPR = M%8
#pragma unroll
  for (int nt = 0; nt < 4; ++nt) {
    int n = nt * 16 + r16;
    float bn = bias[n];
#pragma unroll
    for (int v = 0; v < 8; ++v) {
      int m = m0 + v + 8 * half_id;
      H[(size_t)m * OUT_DIM + n] = acc[nt][v] + bn;
    }
  }
}

// ============================================================================
// K3: init accumulators: out=0, s=0, m=-FLT_MAX (ordered-int)
// ============================================================================
__global__ void k_init(float* __restrict__ out, float* __restrict__ s_sum,
                       int* __restrict__ m_ord, int nNodes) {
  int i = blockIdx.x * blockDim.x + threadIdx.x;
  if (i < nNodes * OUT_DIM) out[i] = 0.0f;
  if (i < nNodes) {
    s_sum[i] = 0.0f;
    m_ord[i] = f2ord(-3.402823466e38f);
  }
}

// ============================================================================
// K4: per-edge attention logit + leaky relu + segment max (wave per edge)
// ============================================================================
__global__ void k_edge_logit(const long long* __restrict__ src,
                             const long long* __restrict__ dst,
                             const float* __restrict__ H,
                             const float* __restrict__ att_w,  // [128]
                             const float* __restrict__ att_b,  // [1]
                             float* __restrict__ e_buf, int* __restrict__ m_ord,
                             int nEdges) {
  const int lane = threadIdx.x & 31;
  const int wid  = blockIdx.x * (blockDim.x >> 5) + (threadIdx.x >> 5);
  if (wid >= nEdges) return;
  const long long s = src[wid], d = dst[wid];
  const float2 hs = ((const float2*)(H + (size_t)s * OUT_DIM))[lane];
  const float2 hd = ((const float2*)(H + (size_t)d * OUT_DIM))[lane];
  const float2 a1 = ((const float2*)att_w)[lane];
  const float2 a2 = ((const float2*)(att_w + OUT_DIM))[lane];
  float p = hs.x * a1.x + hs.y * a1.y + hd.x * a2.x + hd.y * a2.y;
  p = wave_sum32(p);
  if (lane == 0) {
    float e = p + att_b[0];
    e = (e >= 0.0f) ? e : NEG_SLOPE * e;
    e_buf[wid] = e;
    atomicMax(&m_ord[s], f2ord(e));
  }
}

// ============================================================================
// K5: per-edge exp(e - m[src]) and segment sum (thread per edge)
// ============================================================================
__global__ void k_edge_exp(const long long* __restrict__ src,
                           const int* __restrict__ m_ord,
                           float* __restrict__ e_buf, float* __restrict__ s_sum,
                           int nEdges) {
  int i = blockIdx.x * blockDim.x + threadIdx.x;
  if (i >= nEdges) return;
  long long s = src[i];
  float ex = expf(e_buf[i] - ord2f(m_ord[s]));
  e_buf[i] = ex;
  atomicAdd(&s_sum[s], ex);
}

// ============================================================================
// K6: out[dst] += h[src] * alpha   (wave per edge, 2 dims per lane)
// ============================================================================
__global__ void k_scatter(const long long* __restrict__ src,
                          const long long* __restrict__ dst,
                          const float* __restrict__ H,
                          const float* __restrict__ e_buf,
                          const float* __restrict__ s_sum,
                          float* __restrict__ out, int nEdges) {
  const int lane = threadIdx.x & 31;
  const int wid  = blockIdx.x * (blockDim.x >> 5) + (threadIdx.x >> 5);
  if (wid >= nEdges) return;
  const long long s = src[wid], d = dst[wid];
  const float alpha = e_buf[wid] / s_sum[s];
  const float2 hs = ((const float2*)(H + (size_t)s * OUT_DIM))[lane];
  float* o = out + (size_t)d * OUT_DIM + lane * 2;
  atomicAdd(o + 0, hs.x * alpha);
  atomicAdd(o + 1, hs.y * alpha);
}

// ============================================================================
// K7: in-place expmap0 on out (wave per row, 2 elems per lane)
// ============================================================================
__global__ void k_expmap(float* __restrict__ out, int nRows) {
  const int lane = threadIdx.x & 31;
  const int wid  = blockIdx.x * (blockDim.x >> 5) + (threadIdx.x >> 5);
  if (wid >= nRows) return;
  float2* row = (float2*)(out + (size_t)wid * OUT_DIM);
  float2 v = row[lane];
  float ss = wave_sum32(v.x * v.x + v.y * v.y);
  float n = fmaxf(sqrtf(ss), EPS);
  float scale = tanhf(n) / n;
  v.x *= scale; v.y *= scale;
  row[lane] = v;
}

// ============================================================================
extern "C" void kernel_launch(void* const* d_in, const int* in_sizes, int n_in,
                              void* d_out, int out_size, void* d_ws, size_t ws_size,
                              hipStream_t stream) {
  const float*     x     = (const float*)d_in[0];
  const long long* ei    = (const long long*)d_in[1];   // int64 [2, E]
  const float*     W     = (const float*)d_in[2];
  const float*     b     = (const float*)d_in[3];
  const float*     att_w = (const float*)d_in[4];
  const float*     att_b = (const float*)d_in[5];
  float*           out   = (float*)d_out;

  const int N = in_sizes[0] / IN_DIM;       // 100000
  const int E = in_sizes[1] / 2;            // 1600000
  const long long* src = ei;
  const long long* dst = ei + E;

  // workspace carve-up (256B aligned chunks)
  char* ws = (char*)d_ws;
  size_t off = 0;
  auto carve = [&](size_t bytes) { void* p = ws + off; off += (bytes + 255) & ~(size_t)255; return p; };
  _Float16* A16   = (_Float16*)carve((size_t)N * IN_DIM * sizeof(_Float16));
  _Float16* Wt    = (_Float16*)carve((size_t)OUT_DIM * IN_DIM * sizeof(_Float16));
  float*    H     = (float*)   carve((size_t)N * OUT_DIM * sizeof(float));
  float*    e_buf = (float*)   carve((size_t)E * sizeof(float));
  float*    s_sum = (float*)   carve((size_t)N * sizeof(float));
  int*      m_ord = (int*)     carve((size_t)N * sizeof(int));
  (void)ws_size; (void)n_in; (void)out_size;

  const int TPB = 256;                 // 8 waves per block
  const int WPB = TPB / 32;

  // K1: logmap0 -> f16, W transpose
  k_prep<<<512, TPB, 0, stream>>>(x, W, A16, Wt, N);

  // K2: WMMA GEMM (one wave per 16-row strip)
  int nStrips = (N + 15) / 16;
  k_gemm_wmma<<<(nStrips + WPB - 1) / WPB, TPB, 0, stream>>>(A16, Wt, b, H, N);

  // K3: init accumulators + zero output
  int initElems = N * OUT_DIM;
  k_init<<<(initElems + TPB - 1) / TPB, TPB, 0, stream>>>(out, s_sum, m_ord, N);

  // K4: edge logits + segment max
  k_edge_logit<<<(E + WPB - 1) / WPB, TPB, 0, stream>>>(src, dst, H, att_w, att_b,
                                                        e_buf, m_ord, E);
  // K5: exp + segment sum
  k_edge_exp<<<(E + TPB - 1) / TPB, TPB, 0, stream>>>(src, m_ord, e_buf, s_sum, E);

  // K6: weighted scatter
  k_scatter<<<(E + WPB - 1) / WPB, TPB, 0, stream>>>(src, dst, H, e_buf, s_sum, out, E);

  // K7: expmap0 in place
  k_expmap<<<(N + WPB - 1) / WPB, TPB, 0, stream>>>(out, N);
}
